// FiBiNET_18373870092766
// MI455X (gfx1250) — matrix operand order repaired and checked by reading, not compiled, not run on hardware
//
#include <hip/hip_runtime.h>
#include <math.h>

typedef __bf16 bf16_t;
typedef __attribute__((ext_vector_type(16))) __bf16 v16bf;
typedef __attribute__((ext_vector_type(8)))  float  v8f;
typedef int v4i __attribute__((vector_size(16)));
typedef __attribute__((address_space(1))) v4i v4i_as1;
typedef __attribute__((address_space(3))) v4i v4i_as3;

#define B_SAMPLES 4096
#define DNN_K     15424   // 2*120*64 + 64

// ---- async Global->LDS path detection (probe via #warning in stderr) ----
#if defined(__has_builtin)
# if __has_builtin(__builtin_amdgcn_global_load_async_to_lds_b128)
#  define HAVE_ASYNC 1
# endif
#endif
#ifndef HAVE_ASYNC
# define HAVE_ASYNC 0
# warning "CDNA5 probe: __builtin_amdgcn_global_load_async_to_lds_b128 NOT available; using sync LDS staging"
#endif

#if defined(__has_builtin)
# if __has_builtin(__builtin_amdgcn_s_wait_asynccnt)
#  define HAVE_WAIT_ASYNC 1
# endif
#endif
#ifndef HAVE_WAIT_ASYNC
# define HAVE_WAIT_ASYNC 0
# if HAVE_ASYNC
#  warning "CDNA5 probe: s_wait_asynccnt builtin NOT available; using inline asm"
# endif
#endif

__device__ __forceinline__ void wait_async_zero(){
#if HAVE_WAIT_ASYNC
  __builtin_amdgcn_s_wait_asynccnt(0);
#elif HAVE_ASYNC
  asm volatile("s_wait_asynccnt 0x0" ::: "memory");
#endif
}

__device__ __forceinline__ bf16_t f2bf(float x){ return (bf16_t)x; }
__device__ __forceinline__ float  bf2f(bf16_t x){ return (float)x; }

__device__ __forceinline__ v8f wmma_bf16(v16bf a, v16bf b, v8f c){
  return __builtin_amdgcn_wmma_f32_16x16x32_bf16(false, a, false, b, (short)0, c, false, false);
}

// ---------------- generic f32 -> bf16 convert ----------------
__global__ void cvt_f32_bf16(const float* __restrict__ in, bf16_t* __restrict__ out, int n){
  int i = blockIdx.x*blockDim.x + threadIdx.x;
  if(i < n) out[i] = f2bf(in[i]);
}

// ---------------- per-field embedding GEMM --------------------
__global__ __launch_bounds__(128) void emb_gemm(
    const float* __restrict__ sparse, const bf16_t* __restrict__ embW,
    bf16_t* __restrict__ emb)
{
  const int g    = blockIdx.y;
  const int lane = threadIdx.x & 31, wave = threadIdx.x >> 5;
  const int row0 = blockIdx.x*64 + wave*16;
  const int m = lane & 15, hi = lane >> 4;
  const float*  pa = sparse + (size_t)(row0+m)*1024 + g*64 + hi*8;
  const bf16_t* pw = embW + g*4096 + m*64 + hi*8;
  v8f c[4] = {};
  for(int k=0;k<64;k+=32){
    v16bf a;
#pragma unroll
    for(int i=0;i<8;i++){ a[i]=f2bf(pa[k+i]); a[8+i]=f2bf(pa[k+16+i]); }
#pragma unroll
    for(int nf=0;nf<4;nf++){
      const bf16_t* pb = pw + nf*1024;
      v16bf b;
#pragma unroll
      for(int i=0;i<8;i++){ b[i]=pb[k+i]; b[8+i]=pb[k+16+i]; }
      c[nf] = wmma_bf16(a,b,c[nf]);
    }
  }
  const int n = lane & 15, mb = hi*8;
#pragma unroll
  for(int nf=0;nf<4;nf++)
#pragma unroll
    for(int r=0;r<8;r++)
      emb[(size_t)(row0+mb+r)*1024 + g*64 + nf*16 + n] = f2bf(c[nf][r]);
}

// ---------------- SENet scalars -------------------------------
__global__ __launch_bounds__(256) void senet_kernel(
    const bf16_t* __restrict__ emb, const float* __restrict__ seW1,
    const float* __restrict__ seW2, float* __restrict__ a_se)
{
  int b = blockIdx.x*blockDim.x + threadIdx.x;
  if(b >= B_SAMPLES) return;
  const bf16_t* e = emb + (size_t)b*1024;
  float z[16];
#pragma unroll
  for(int g=0; g<16; g++){
    float s = 0.f;
    for(int k=0;k<64;k++) s += bf2f(e[g*64+k]);
    z[g] = s * (1.f/64.f);
  }
  float t[4];
#pragma unroll
  for(int r=0;r<4;r++){
    float s = 0.f;
#pragma unroll
    for(int g=0;g<16;g++) s += z[g]*seW1[r*16+g];
    t[r] = fmaxf(s, 0.f);
  }
#pragma unroll
  for(int g=0;g<16;g++){
    float s = 0.f;
#pragma unroll
    for(int r=0;r<4;r++) s += t[r]*seW2[g*4+r];
    a_se[b*16+g] = fmaxf(s, 0.f);
  }
}

// ---------------- bilinear pairs + DNN input assembly ---------
__global__ __launch_bounds__(128) void bilinear_kernel(
    const bf16_t* __restrict__ emb, const bf16_t* __restrict__ biW,
    const float* __restrict__ a_se, bf16_t* __restrict__ dnn)
{
  const int p = blockIdx.y;
  int fi = 0, acc = 0;
  while(acc + (15-fi) <= p){ acc += 15-fi; fi++; }
  const int fj = fi + 1 + (p - acc);

  const int lane = threadIdx.x & 31, wave = threadIdx.x >> 5;
  const int row0 = blockIdx.x*64 + wave*16;
  const int m = lane & 15, hi = lane >> 4;
  const bf16_t* pa = emb + (size_t)(row0+m)*1024 + fi*64 + hi*8;
  const bf16_t* pw = biW + p*4096 + m*64 + hi*8;
  v8f c[4] = {};
  for(int k=0;k<64;k+=32){
    v16bf a;
#pragma unroll
    for(int i=0;i<8;i++){ a[i]=pa[k+i]; a[8+i]=pa[k+16+i]; }
#pragma unroll
    for(int nf=0;nf<4;nf++){
      const bf16_t* pb = pw + nf*1024;
      v16bf b;
#pragma unroll
      for(int i=0;i<8;i++){ b[i]=pb[k+i]; b[8+i]=pb[k+16+i]; }
      c[nf] = wmma_bf16(a,b,c[nf]);
    }
  }
  const int n = lane & 15, mb = hi*8;
#pragma unroll
  for(int r=0;r<8;r++){
    const int row = row0+mb+r;
    const float aij = a_se[row*16+fi]*a_se[row*16+fj];
    const bf16_t* ej = emb + (size_t)row*1024 + fj*64;
    bf16_t* dp = dnn + (size_t)row*DNN_K + p*64;
#pragma unroll
    for(int nf=0;nf<4;nf++){
      const int f = nf*16+n;
      const float v = c[nf][r]*bf2f(ej[f]);
      dp[f]      = f2bf(v);
      dp[7680+f] = f2bf(v*aij);
    }
  }
}

__global__ void dense_copy(const float* __restrict__ dense, bf16_t* __restrict__ dnn){
  int i = blockIdx.x*blockDim.x + threadIdx.x;
  if(i >= B_SAMPLES*64) return;
  int b = i >> 6, d = i & 63;
  dnn[(size_t)b*DNN_K + 15360 + d] = f2bf(dense[i]);
}

// ---------------- big WMMA GEMM: C = A(MxK) * W(NxK)^T + bias --
// block = 8 waves (2x4); block tile 64x256; wave tile 32x64 (2x4 frags).
// B block tile (256 rows x 32 K, bf16) staged in LDS via async DMA,
// double-buffered so the copy overlaps WMMA. Row stride padded to 40
// elements (80B) => conflict-free b128 LDS reads.
#define LDSB_STRIDE 40
__global__ __launch_bounds__(256) void gemm_bf16_nt(
    const bf16_t* __restrict__ A, int lda,
    const bf16_t* __restrict__ W, int ldb,
    const float* __restrict__ bias,
    float* __restrict__ C, int ldc, int K)
{
  __shared__ bf16_t ldsB[2][256*LDSB_STRIDE];   // 2 x 20KB
  const int tid  = threadIdx.x;
  const int lane = tid & 31;
  const int wave = tid >> 5;
  const int wm = wave >> 2, wn = wave & 3;
  const int row0 = blockIdx.x*64 + wm*32;
  const int colB = blockIdx.y*256;
  const int col0 = colB + wn*64;
  const int m = lane & 15, hi = lane >> 4;

  const bf16_t* pa0 = A + (size_t)(row0+m)*lda + hi*8;
  const bf16_t* pa1 = pa0 + (size_t)16*lda;
  const bf16_t* pw  = W + (size_t)(colB + tid)*ldb;   // this thread stages weight row colB+tid

  auto stage = [&](int bufsel, int k){
    bf16_t* dst = &ldsB[bufsel][tid*LDSB_STRIDE];
    const bf16_t* src = pw + k;
#if HAVE_ASYNC
#pragma unroll
    for(int c=0;c<4;c++)
      __builtin_amdgcn_global_load_async_to_lds_b128(
          (v4i_as1*)(src + c*8), (v4i_as3*)(dst + c*8), 0, 0);
#else
#pragma unroll
    for(int c=0;c<4;c++)
      *(v4i*)(dst + c*8) = *(const v4i*)(src + c*8);
#endif
  };

  v8f acc[2][4] = {};
  stage(0, 0);
  wait_async_zero();
  __syncthreads();
  int buf = 0;
  for(int k=0;k<K;k+=32){
    if(k+32 < K) stage(buf^1, k+32);
    __builtin_prefetch(pa0 + k + 2048, 0, 1);   // global_prefetch_b8
    v16bf a0,a1;
#pragma unroll
    for(int i=0;i<8;i++){
      a0[i]=pa0[k+i];  a0[8+i]=pa0[k+16+i];
      a1[i]=pa1[k+i];  a1[8+i]=pa1[k+16+i];
    }
#pragma unroll
    for(int nf=0;nf<4;nf++){
      const bf16_t* lb = &ldsB[buf][(wn*64 + nf*16 + m)*LDSB_STRIDE + hi*8];
      v16bf b;
#pragma unroll
      for(int i=0;i<8;i++){ b[i]=lb[i]; b[8+i]=lb[16+i]; }
      acc[0][nf] = wmma_bf16(a0,b,acc[0][nf]);
      acc[1][nf] = wmma_bf16(a1,b,acc[1][nf]);
    }
    wait_async_zero();
    __syncthreads();
    buf ^= 1;
  }

  const int n = lane & 15, mb = hi*8;
#pragma unroll
  for(int nf=0;nf<4;nf++){
    const int cc = col0 + nf*16 + n;
    const float bb = bias ? bias[cc] : 0.f;
#pragma unroll
    for(int r=0;r<8;r++){
      C[(size_t)(row0+mb+r)*ldc + cc]    = acc[0][nf][r]+bb;
      C[(size_t)(row0+16+mb+r)*ldc + cc] = acc[1][nf][r]+bb;
    }
  }
}

// ---------------- batchnorm stats over batch axis -------------
__global__ __launch_bounds__(256) void colstats(
    const float* __restrict__ Y, int M, int N,
    float* __restrict__ mean, float* __restrict__ rstd)
{
  __shared__ float sh[256], sh2[256];
  const int n = blockIdx.x;
  float s=0.f, s2=0.f;
  for(int r=threadIdx.x; r<M; r+=256){
    float v = Y[(size_t)r*N + n]; s += v; s2 += v*v;
  }
  sh[threadIdx.x]=s; sh2[threadIdx.x]=s2; __syncthreads();
  for(int off=128; off>0; off>>=1){
    if(threadIdx.x < off){ sh[threadIdx.x]+=sh[threadIdx.x+off]; sh2[threadIdx.x]+=sh2[threadIdx.x+off]; }
    __syncthreads();
  }
  if(threadIdx.x==0){
    float mu  = sh[0]/M;
    float var = sh2[0]/M - mu*mu;
    mean[n] = mu; rstd[n] = rsqrtf(var + 1e-5f);
  }
}

__global__ void norm_sigmoid(const float* __restrict__ Y,
                             const float* __restrict__ mean, const float* __restrict__ rstd,
                             bf16_t* __restrict__ H, int N, int total)
{
  int i = blockIdx.x*blockDim.x + threadIdx.x;
  if(i >= total) return;
  int n = i % N;
  float v = (Y[i]-mean[n])*rstd[n];
  H[i] = f2bf(1.f/(1.f+expf(-v)));
}

// ---------------- tail kernels --------------------------------
__global__ void linear_logit(const float* __restrict__ sparse, const float* __restrict__ dense,
                             const float* __restrict__ linW, const float* __restrict__ linB,
                             float* __restrict__ ll)
{
  int b = blockIdx.x*blockDim.x + threadIdx.x;
  if(b >= B_SAMPLES) return;
  float s = linB[0];
  const float* sp = sparse + (size_t)b*1024;
  for(int k=0;k<1024;k++) s += sp[k]*linW[k];
  const float* dn = dense + (size_t)b*64;
  for(int d=0;d<64;d++) s += dn[d]*linW[1024+d];
  ll[b] = s;
}

__global__ void layer3(const bf16_t* __restrict__ h2, const float* __restrict__ W3,
                       const float* __restrict__ b3, float* __restrict__ z3)
{
  int b = blockIdx.x*blockDim.x + threadIdx.x;
  if(b >= B_SAMPLES) return;
  float s = b3[0];
  const bf16_t* h = h2 + (size_t)b*256;
  for(int k=0;k<256;k++) s += bf2f(h[k])*W3[k];
  z3[b] = s;
}

__global__ void final_kernel(const float* __restrict__ ll, const float* __restrict__ z3,
                             const float* __restrict__ mean, const float* __restrict__ rstd,
                             float* __restrict__ out)
{
  int b = blockIdx.x*blockDim.x + threadIdx.x;
  if(b >= B_SAMPLES) return;
  float d = 1.f/(1.f+expf(-((z3[b]-mean[0])*rstd[0])));
  out[b] = 1.f/(1.f+expf(-(ll[b]+d)));
}

// ---------------- workspace layout (bytes) --------------------
static constexpr size_t OFF_EMBW = 0;
static constexpr size_t OFF_BIW  = OFF_EMBW + 131072;
static constexpr size_t OFF_W0   = OFF_BIW  + 983040;
static constexpr size_t OFF_W1   = OFF_W0   + 31588352;
static constexpr size_t OFF_W2   = OFF_W1   + 1048576;
static constexpr size_t OFF_EMB  = OFF_W2   + 262144;
static constexpr size_t OFF_ASE  = OFF_EMB  + 8388608;
static constexpr size_t OFF_DNN  = OFF_ASE  + 262144;
static constexpr size_t OFF_Y    = OFF_DNN  + 126353408;
static constexpr size_t OFF_H0   = OFF_Y    + 16777216;
static constexpr size_t OFF_H1   = OFF_H0   + 8388608;
static constexpr size_t OFF_H2   = OFF_H1   + 4194304;
static constexpr size_t OFF_MEAN = OFF_H2   + 2097152;
static constexpr size_t OFF_RSTD = OFF_MEAN + 4096;
static constexpr size_t OFF_Z3   = OFF_RSTD + 4096;
static constexpr size_t OFF_LL   = OFF_Z3   + 16384;

extern "C" void kernel_launch(void* const* d_in, const int* in_sizes, int n_in,
                              void* d_out, int out_size, void* d_ws, size_t ws_size,
                              hipStream_t stream)
{
  (void)in_sizes; (void)n_in; (void)out_size; (void)ws_size;
  const float* sparse = (const float*)d_in[0];
  const float* dense  = (const float*)d_in[1];
  const float* embW   = (const float*)d_in[2];
  const float* linW   = (const float*)d_in[3];
  const float* linB   = (const float*)d_in[4];
  const float* seW1   = (const float*)d_in[5];
  const float* seW2   = (const float*)d_in[6];
  const float* biW    = (const float*)d_in[7];
  const float* W0     = (const float*)d_in[8];
  const float* b0     = (const float*)d_in[9];
  const float* W1     = (const float*)d_in[10];
  const float* b1     = (const float*)d_in[11];
  const float* W2     = (const float*)d_in[12];
  const float* b2     = (const float*)d_in[13];
  const float* W3     = (const float*)d_in[14];
  const float* b3     = (const float*)d_in[15];
  float* out = (float*)d_out;

  char* ws = (char*)d_ws;
  bf16_t* embW_bf = (bf16_t*)(ws + OFF_EMBW);
  bf16_t* biW_bf  = (bf16_t*)(ws + OFF_BIW);
  bf16_t* W0_bf   = (bf16_t*)(ws + OFF_W0);
  bf16_t* W1_bf   = (bf16_t*)(ws + OFF_W1);
  bf16_t* W2_bf   = (bf16_t*)(ws + OFF_W2);
  bf16_t* emb_bf  = (bf16_t*)(ws + OFF_EMB);
  float*  a_se    = (float*) (ws + OFF_ASE);
  bf16_t* dnn_bf  = (bf16_t*)(ws + OFF_DNN);
  float*  ybuf    = (float*) (ws + OFF_Y);
  bf16_t* h0_bf   = (bf16_t*)(ws + OFF_H0);
  bf16_t* h1_bf   = (bf16_t*)(ws + OFF_H1);
  bf16_t* h2_bf   = (bf16_t*)(ws + OFF_H2);
  float*  meanb   = (float*) (ws + OFF_MEAN);
  float*  rstdb   = (float*) (ws + OFF_RSTD);
  float*  z3      = (float*) (ws + OFF_Z3);
  float*  ll      = (float*) (ws + OFF_LL);

  // weight conversions
  cvt_f32_bf16<<<(16*64*64+255)/256,    256, 0, stream>>>(embW, embW_bf, 16*64*64);
  cvt_f32_bf16<<<(120*64*64+255)/256,   256, 0, stream>>>(biW,  biW_bf,  120*64*64);
  cvt_f32_bf16<<<(1024*15424+255)/256,  256, 0, stream>>>(W0,   W0_bf,   1024*15424);
  cvt_f32_bf16<<<(512*1024+255)/256,    256, 0, stream>>>(W1,   W1_bf,   512*1024);
  cvt_f32_bf16<<<(256*512+255)/256,     256, 0, stream>>>(W2,   W2_bf,   256*512);

  // features
  emb_gemm<<<dim3(64,16), 128, 0, stream>>>(sparse, embW_bf, emb_bf);
  senet_kernel<<<16, 256, 0, stream>>>(emb_bf, seW1, seW2, a_se);
  bilinear_kernel<<<dim3(64,120), 128, 0, stream>>>(emb_bf, biW_bf, a_se, dnn_bf);
  dense_copy<<<(B_SAMPLES*64+255)/256, 256, 0, stream>>>(dense, dnn_bf);

  // layer 0: (4096x15424) x (1024x15424)^T
  gemm_bf16_nt<<<dim3(64,4), 256, 0, stream>>>(dnn_bf, DNN_K, W0_bf, DNN_K, b0, ybuf, 1024, DNN_K);
  colstats<<<1024, 256, 0, stream>>>(ybuf, B_SAMPLES, 1024, meanb, rstdb);
  norm_sigmoid<<<(B_SAMPLES*1024+255)/256, 256, 0, stream>>>(ybuf, meanb, rstdb, h0_bf, 1024, B_SAMPLES*1024);

  // layer 1
  gemm_bf16_nt<<<dim3(64,2), 256, 0, stream>>>(h0_bf, 1024, W1_bf, 1024, b1, ybuf, 512, 1024);
  colstats<<<512, 256, 0, stream>>>(ybuf, B_SAMPLES, 512, meanb, rstdb);
  norm_sigmoid<<<(B_SAMPLES*512+255)/256, 256, 0, stream>>>(ybuf, meanb, rstdb, h1_bf, 512, B_SAMPLES*512);

  // layer 2
  gemm_bf16_nt<<<dim3(64,1), 256, 0, stream>>>(h1_bf, 512, W2_bf, 512, b2, ybuf, 256, 512);
  colstats<<<256, 256, 0, stream>>>(ybuf, B_SAMPLES, 256, meanb, rstdb);
  norm_sigmoid<<<(B_SAMPLES*256+255)/256, 256, 0, stream>>>(ybuf, meanb, rstdb, h2_bf, 256, B_SAMPLES*256);

  // tail
  layer3<<<16, 256, 0, stream>>>(h2_bf, W3, b3, z3);
  colstats<<<1, 256, 0, stream>>>(z3, B_SAMPLES, 1, meanb, rstdb);
  linear_logit<<<16, 256, 0, stream>>>(sparse, dense, linW, linB, ll);
  final_kernel<<<16, 256, 0, stream>>>(ll, z3, meanb, rstdb, out);
}
